// Rho_6390911336720
// MI455X (gfx1250) — matrix operand order repaired and checked by reading, not compiled, not run on hardware
//
#include <hip/hip_runtime.h>
#include <hip/hip_bf16.h>
#include <math.h>

// ---------------------------------------------------------------------------
// Heterogeneous graph transformer (PyG TransformerConv x2 layers) for gfx1250.
// Dense GEMMs: v_wmma_f32_16x16x32_f16 with async global->LDS double-buffered
// weight staging (ASYNCcnt). Scatter/softmax: wave-per-edge, coalesced 128b
// loads, f32 global atomics.
// ---------------------------------------------------------------------------

typedef __attribute__((ext_vector_type(16))) _Float16 v16h;
typedef __attribute__((ext_vector_type(8)))  _Float16 v8h;
typedef __attribute__((ext_vector_type(8)))  float    v8f;
typedef __attribute__((ext_vector_type(4)))  int      v4i;

#define HIDDEN   256
#define HEADS    4
#define BGRAPH   32

// ---- async global->LDS staging (gfx1250 ASYNCcnt path, with fallback) -----
#if defined(__gfx1250__) && __has_builtin(__builtin_amdgcn_global_load_async_to_lds_b128)
#define HAVE_ASYNC_G2L 1
#else
#define HAVE_ASYNC_G2L 0
#endif

__device__ __forceinline__ void g2l_16B(_Float16* l, const _Float16* g) {
#if HAVE_ASYNC_G2L
  // Builtin signature (probe round 3): (int4 AS1*, int4 AS3*, imm offset, imm cpol)
  __builtin_amdgcn_global_load_async_to_lds_b128(
      (__attribute__((address_space(1))) v4i*)g,
      (__attribute__((address_space(3))) v4i*)l, 0, 0);
#else
  *(v8h*)l = *(const v8h*)g;
#endif
}

#if HAVE_ASYNC_G2L
#  if __has_builtin(__builtin_amdgcn_s_wait_asynccnt)
#    define WAIT_ASYNC(n) __builtin_amdgcn_s_wait_asynccnt(n)
#  else
#    define WAIT_ASYNC(n) asm volatile("s_wait_asynccnt %0" ::"i"(n) : "memory")
#  endif
#else
#  define WAIT_ASYNC(n) ((void)0)
#endif

// ---------------------------------------------------------------------------
// Weight repack: f32 [K,256] row-major -> f16 WMMA-B-fragment-major.
// Fragment layout (ISA 7.12.2, 16-bit B 32x16): lane L holds column N=L&15,
// K = (j>>3)*16 + (L>>4)*8 + (j&7)  for j in 0..15.  Pack index:
//   P[((kt*16 + nt)*32 + lane)*16 + j]
// ---------------------------------------------------------------------------
__global__ void pack_weight_kernel(const float* __restrict__ W,
                                   _Float16* __restrict__ P,
                                   int Ka, int KT) {
  int idx = blockIdx.x * 256 + threadIdx.x;
  int total = KT * 8192;              // KT*16*32*16
  if (idx >= total) return;
  int j    = idx & 15;
  int lane = (idx >> 4) & 31;
  int nt   = (idx >> 9) & 15;
  int kt   = idx >> 13;
  int k = kt * 32 + ((j >> 3) << 4) + ((lane >> 4) << 3) + (j & 7);
  int n = nt * 16 + (lane & 15);
  P[idx] = (k < Ka) ? (_Float16)W[(size_t)k * HIDDEN + n] : (_Float16)0.0f;
}

// ---------------------------------------------------------------------------
// GEMM: C[M,256] = act(A[M,Ka] @ W[Ka,256] + bias)
// 256 threads = 8 waves; wave = (row-tile mt 0..3, N-half nh 0..1);
// each wave: 16 rows x 128 cols = 8 WMMA accumulators (64 VGPRs).
// Packed B fragments are double-buffered through LDS (16KB per K-chunk)
// using async global->LDS copies overlapped with the WMMAs of the current
// chunk; per-chunk handshake = s_wait_asynccnt + workgroup barrier.
// ---------------------------------------------------------------------------
template <bool FULLK>
__device__ __forceinline__ void gemm_body(
    const float* __restrict__ A, const _Float16* __restrict__ Wp,
    const float* __restrict__ bias, float* __restrict__ Cf,
    _Float16* __restrict__ Ch, int M, int Ka, int KT, int act,
    _Float16 (*lbuf)[8192]) {
  int tid   = threadIdx.x;
  int wave  = tid >> 5;
  int lane  = tid & 31;
  int lmod  = lane & 15;
  int lhalf = lane >> 4;
  int mt = wave >> 1;                 // row tile within block (0..3)
  int nh = wave & 1;                  // N half (0..1), 128 cols each
  int rowBase = blockIdx.x * 64 + mt * 16;

  const float* arow = A + (size_t)(rowBase + lmod) * Ka;

  v8f zero = {};
  v8f acc[8];
#pragma unroll
  for (int t = 0; t < 8; ++t) acc[t] = zero;

  // stage K-chunk 0 (16 tiles x 512 halves = 16KB; 4 x 16B per thread)
  {
    const _Float16* s0 = Wp;
#pragma unroll
    for (int i = 0; i < 4; ++i) {
      int off = (i * 256 + tid) * 8;
      g2l_16B(&lbuf[0][off], s0 + off);
    }
  }

  for (int kt = 0; kt < KT; ++kt) {
    int cur = kt & 1;
    if (kt + 1 < KT) {                // prefetch next chunk into other buffer
      const _Float16* sn = Wp + (size_t)(kt + 1) * 8192;
#pragma unroll
      for (int i = 0; i < 4; ++i) {
        int off = (i * 256 + tid) * 8;
        g2l_16B(&lbuf[cur ^ 1][off], sn + off);
      }
      WAIT_ASYNC(4);                  // current chunk's 4 copies complete
    } else {
      WAIT_ASYNC(0);
    }
    __syncthreads();

    // A fragment (ISA 7.12.2, 16-bit A 16x32): lane L (row M=L&15) holds
    // K = (j>>3)*16 + (L>>4)*8 + (j&7).
    v16h a;
    if (FULLK) {
      const float4* a0 = (const float4*)(arow + kt * 32 + (lhalf << 3));
      const float4* a1 = (const float4*)(arow + kt * 32 + 16 + (lhalf << 3));
      float4 f0 = a0[0], f1 = a0[1], f2 = a1[0], f3 = a1[1];
      a[0]  = (_Float16)f0.x; a[1]  = (_Float16)f0.y;
      a[2]  = (_Float16)f0.z; a[3]  = (_Float16)f0.w;
      a[4]  = (_Float16)f1.x; a[5]  = (_Float16)f1.y;
      a[6]  = (_Float16)f1.z; a[7]  = (_Float16)f1.w;
      a[8]  = (_Float16)f2.x; a[9]  = (_Float16)f2.y;
      a[10] = (_Float16)f2.z; a[11] = (_Float16)f2.w;
      a[12] = (_Float16)f3.x; a[13] = (_Float16)f3.y;
      a[14] = (_Float16)f3.z; a[15] = (_Float16)f3.w;
    } else {
#pragma unroll
      for (int j = 0; j < 16; ++j) {
        int kk = kt * 32 + ((j >> 3) << 4) + (lhalf << 3) + (j & 7);
        a[j] = (kk < Ka) ? (_Float16)arow[kk] : (_Float16)0.0f;
      }
    }

    // 8 B fragments from LDS (ds_load_b128 pairs), then 8 independent WMMAs.
    const _Float16* bb = &lbuf[cur][(nh * 8) * 512 + lane * 16];
    v16h b[8];
#pragma unroll
    for (int t = 0; t < 8; ++t)
      b[t] = *(const v16h*)(bb + t * 512);
#pragma unroll
    for (int t = 0; t < 8; ++t)
      acc[t] = __builtin_amdgcn_wmma_f32_16x16x32_f16(
          false, a, false, b[t], (short)0, acc[t], false, false);

    __syncthreads();                  // protect buffer about to be re-staged
  }

  // Epilogue: D layout = lane N=L&15; VGPR i -> M = i + 8*(L>>4).
#pragma unroll
  for (int t = 0; t < 8; ++t) {
    int n = (nh * 8 + t) * 16 + lmod;
    float bv = bias ? bias[n] : 0.0f;
#pragma unroll
    for (int i = 0; i < 8; ++i) {
      int mr = rowBase + (lhalf << 3) + i;
      float val = acc[t][i] + bv;
      if (act == 1) val = val / (1.0f + __expf(-val));   // SiLU
      if (Ch) Ch[(size_t)mr * HIDDEN + n] = (_Float16)val;
      else    Cf[(size_t)mr * HIDDEN + n] = val;
    }
  }
}

__global__ void __launch_bounds__(256)
gemm_wmma_kernel(const float* __restrict__ A, const _Float16* __restrict__ Wp,
                 const float* __restrict__ bias, float* __restrict__ Cf,
                 _Float16* __restrict__ Ch, int M, int Ka, int KT, int act) {
  __shared__ _Float16 lbuf[2][8192];    // 2 x 16KB double buffer
  if (Ka == KT * 32)
    gemm_body<true>(A, Wp, bias, Cf, Ch, M, Ka, KT, act, lbuf);
  else
    gemm_body<false>(A, Wp, bias, Cf, Ch, M, Ka, KT, act, lbuf);
}

// ---------------------------------------------------------------------------
// Edge attention: wave per edge, lane owns 8 contiguous channels.
// score[e,h] = dot(q[dst], k[src]+e_proj) / sqrt(64); segment max via
// order-preserving uint encoding + atomicMax (global_atomic_max_u32).
// ---------------------------------------------------------------------------
__device__ __forceinline__ unsigned enc_f32(float f) {
  unsigned b = __float_as_uint(f);
  return (b & 0x80000000u) ? ~b : (b | 0x80000000u);
}
__device__ __forceinline__ float dec_f32(unsigned u) {
  return __uint_as_float((u & 0x80000000u) ? (u & 0x7FFFFFFFu) : ~u);
}

__global__ void edge_score_kernel(const float* __restrict__ q,
                                  const float* __restrict__ k,
                                  const _Float16* __restrict__ e16,
                                  const int* __restrict__ src,
                                  const int* __restrict__ dst,
                                  float* __restrict__ score,
                                  unsigned* __restrict__ segmax,
                                  int E) {
  int gtid = blockIdx.x * blockDim.x + threadIdx.x;
  int e = gtid >> 5;
  int lane = threadIdx.x & 31;
  if (e >= E) return;
  int s = src[e], d = dst[e];
  const float*    qr = q   + (size_t)d * HIDDEN + lane * 8;
  const float*    kr = k   + (size_t)s * HIDDEN + lane * 8;
  const _Float16* er = e16 + (size_t)e * HIDDEN + lane * 8;
  float part = 0.0f;
#pragma unroll
  for (int c = 0; c < 8; ++c) part += qr[c] * (kr[c] + (float)er[c]);
  // reduce within each 8-lane head group (head = 64 channels)
  part += __shfl_xor(part, 1, 32);
  part += __shfl_xor(part, 2, 32);
  part += __shfl_xor(part, 4, 32);
  if ((lane & 7) == 0) {
    int h = lane >> 3;
    float sc = part * 0.125f;                 // 1/sqrt(HEAD_DIM)
    score[(size_t)e * HEADS + h] = sc;
    atomicMax(&segmax[(size_t)d * HEADS + h], enc_f32(sc));
  }
}

__global__ void edge_exp_kernel(float* __restrict__ score,
                                const int* __restrict__ dst,
                                const unsigned* __restrict__ segmax,
                                float* __restrict__ segden,
                                int E) {
  int idx = blockIdx.x * blockDim.x + threadIdx.x;
  if (idx >= E * HEADS) return;
  int e = idx >> 2, h = idx & 3;
  int d = dst[e];
  float m  = dec_f32(segmax[(size_t)d * HEADS + h]);
  float ex = __expf(score[idx] - m);
  score[idx] = ex;
  atomicAdd(&segden[(size_t)d * HEADS + h], ex);
}

__global__ void edge_msg_kernel(const float* __restrict__ score,
                                const float* __restrict__ segden,
                                const float* __restrict__ v,
                                const _Float16* __restrict__ e16,
                                const int* __restrict__ src,
                                const int* __restrict__ dst,
                                float* __restrict__ out,
                                int E) {
  int gtid = blockIdx.x * blockDim.x + threadIdx.x;
  int e = gtid >> 5;
  int lane = threadIdx.x & 31;
  if (e >= E) return;
  int s = src[e], d = dst[e];
  int h = lane >> 3;
  float den   = segden[(size_t)d * HEADS + h];
  float alpha = score[(size_t)e * HEADS + h] / fmaxf(den, 1e-16f);
  const float*    vr = v   + (size_t)s * HIDDEN + lane * 8;
  const _Float16* er = e16 + (size_t)e * HIDDEN + lane * 8;
  float* orow = out + (size_t)d * HIDDEN + lane * 8;
#pragma unroll
  for (int c = 0; c < 8; ++c)
    atomicAdd(&orow[c], (vr[c] + (float)er[c]) * alpha);
}

// ---------------------------------------------------------------------------
// LayerNorm + SiLU, wave per row (256 channels, lane owns 8).
// ---------------------------------------------------------------------------
__global__ void ln_silu_kernel(const float* __restrict__ in,
                               const float* __restrict__ g,
                               const float* __restrict__ b,
                               float* __restrict__ out, int N) {
  int gtid = blockIdx.x * blockDim.x + threadIdx.x;
  int row = gtid >> 5;
  int lane = threadIdx.x & 31;
  if (row >= N) return;
  const float* r = in + (size_t)row * HIDDEN + lane * 8;
  float vv[8], s = 0.0f, s2 = 0.0f;
#pragma unroll
  for (int c = 0; c < 8; ++c) { vv[c] = r[c]; s += vv[c]; s2 += vv[c] * vv[c]; }
#pragma unroll
  for (int off = 1; off < 32; off <<= 1) {
    s  += __shfl_xor(s,  off, 32);
    s2 += __shfl_xor(s2, off, 32);
  }
  float mu  = s * (1.0f / HIDDEN);
  float var = s2 * (1.0f / HIDDEN) - mu * mu;
  float rs  = rsqrtf(var + 1e-5f);
  float* o = out + (size_t)row * HIDDEN + lane * 8;
#pragma unroll
  for (int c = 0; c < 8; ++c) {
    int ch = lane * 8 + c;
    float y = (vv[c] - mu) * rs * g[ch] + b[ch];
    o[c] = y / (1.0f + __expf(-y));   // SiLU
  }
}

// ---------------------------------------------------------------------------
// Mean pool: wave per node accumulates into per-graph sums; finalize divides
// and builds the concatenated pooled output.
// ---------------------------------------------------------------------------
__global__ void pool_acc_kernel(const float* __restrict__ node,
                                const int* __restrict__ batch,
                                float* __restrict__ sum,
                                float* __restrict__ cnt, int N) {
  int gtid = blockIdx.x * blockDim.x + threadIdx.x;
  int n = gtid >> 5;
  int lane = threadIdx.x & 31;
  if (n >= N) return;
  int bidx = batch[n];
  const float* r = node + (size_t)n * HIDDEN + lane * 8;
  float* srow = sum + (size_t)bidx * HIDDEN + lane * 8;
#pragma unroll
  for (int c = 0; c < 8; ++c) atomicAdd(&srow[c], r[c]);
  if (lane == 0) atomicAdd(&cnt[bidx], 1.0f);
}

__global__ void pool_final_kernel(float* __restrict__ pa, float* __restrict__ ps,
                                  const float* __restrict__ cntA,
                                  const float* __restrict__ cntS,
                                  float* __restrict__ concat) {
  int idx = blockIdx.x * 256 + threadIdx.x;
  if (idx >= BGRAPH * HIDDEN) return;
  int bg = idx >> 8, c = idx & 255;
  float va = pa[idx] / fmaxf(cntA[bg], 1.0f);
  float vs = ps[idx] / fmaxf(cntS[bg], 1.0f);
  pa[idx] = va;
  ps[idx] = vs;
  concat[(size_t)bg * 512 + c]       = va;
  concat[(size_t)bg * 512 + 256 + c] = vs;
}

// ---------------------------------------------------------------------------
// Host orchestration
// ---------------------------------------------------------------------------
static void launch_gemm(hipStream_t s, const float* A, const _Float16* Wp,
                        const float* bias, float* Cf, _Float16* Ch,
                        int M, int Ka, int KT, int act) {
  gemm_wmma_kernel<<<dim3(M / 64), dim3(256), 0, s>>>(A, Wp, bias, Cf, Ch, M, Ka, KT, act);
}

// d_in flattening (setup_inputs dict order; params as a JAX pytree with
// dict keys sorted alphabetically at each level):
//  0 x_agent  1 x_scene  2 ea_as  3 ea_sa  4 src_as  5 dst_as  6 src_sa
//  7 dst_sa  8 batch_agent  9 batch_scene
// 10 in_agent.b 11 in_agent.w 12 in_scene.b 13 in_scene.w
// per-conv block (e.w, k.b, k.w, q.b, q.w, skip.b, skip.w, v.b, v.w):
//   L0.as @14, L0.sa @23, L1.as @32, L1.sa @41
// 50 norm_agent.b 51 norm_agent.g 52 norm_scene.b 53 norm_scene.g
// 54 out_agent.b 55 out_agent.w 56 out_scene.b 57 out_scene.w
#define CV_EW 0
#define CV_KB 1
#define CV_KW 2
#define CV_QB 3
#define CV_QW 4
#define CV_SKB 5
#define CV_SKW 6
#define CV_VB 7
#define CV_VW 8

extern "C" void kernel_launch(void* const* d_in, const int* in_sizes, int n_in,
                              void* d_out, int out_size, void* d_ws, size_t ws_size,
                              hipStream_t stream) {
  const int NA = in_sizes[0] / 16;
  const int NS = in_sizes[1] / 16;
  const int E  = in_sizes[4];

  const float* x_agent = (const float*)d_in[0];
  const float* x_scene = (const float*)d_in[1];
  const float* ea_as   = (const float*)d_in[2];
  const float* ea_sa   = (const float*)d_in[3];
  const int* src_as = (const int*)d_in[4];
  const int* dst_as = (const int*)d_in[5];
  const int* src_sa = (const int*)d_in[6];
  const int* dst_sa = (const int*)d_in[7];
  const int* batch_a = (const int*)d_in[8];
  const int* batch_s = (const int*)d_in[9];

  // ---- workspace carve-up -------------------------------------------------
  char* p = (char*)d_ws;
  auto alloc = [&](size_t bytes) -> void* {
    void* r = (void*)p;
    p += (bytes + 255) & ~(size_t)255;
    return r;
  };
  float*    xa     = (float*)alloc((size_t)NA * HIDDEN * 4);
  float*    xs     = (float*)alloc((size_t)NS * HIDDEN * 4);
  float*    qbuf   = (float*)alloc((size_t)NS * HIDDEN * 4);
  float*    kbuf   = (float*)alloc((size_t)NS * HIDDEN * 4);
  float*    vbuf   = (float*)alloc((size_t)NS * HIDDEN * 4);
  float*    new_a  = (float*)alloc((size_t)NA * HIDDEN * 4);
  float*    new_s  = (float*)alloc((size_t)NS * HIDDEN * 4);
  float*    score  = (float*)alloc((size_t)E * HEADS * 4);
  unsigned* segmax = (unsigned*)alloc((size_t)NS * HEADS * 4);
  float*    segden = (float*)alloc((size_t)NS * HEADS * 4);
  float*    cnt    = (float*)alloc(2 * BGRAPH * 4);       // [cntA | cntS]
  _Float16* e16    = (_Float16*)alloc((size_t)E * HIDDEN * 2);
  // packed weights: in(2x 32*256) + 4 convs (4x 256*256 + 64*256) + out(2x 256*256)
  const size_t PK_CONV = 4 * 65536 + 16384;
  _Float16* wpack  = (_Float16*)alloc((2 * 8192 + 4 * PK_CONV + 2 * 65536) * 2);

  _Float16* pk_inA  = wpack;
  _Float16* pk_inS  = wpack + 8192;
  _Float16* pk_conv = wpack + 16384;                       // 4 conv blocks
  _Float16* pk_outA = pk_conv + 4 * PK_CONV;
  _Float16* pk_outS = pk_outA + 65536;

  // ---- pack all weights to f16 WMMA fragments -----------------------------
  auto pack = [&](const float* W, _Float16* P, int Ka, int KT) {
    int total = KT * 8192;
    pack_weight_kernel<<<dim3((total + 255) / 256), dim3(256), 0, stream>>>(W, P, Ka, KT);
  };
  pack((const float*)d_in[11], pk_inA, 16, 1);
  pack((const float*)d_in[13], pk_inS, 16, 1);
  const int convBase[4] = {14, 23, 32, 41};                // L0.as, L0.sa, L1.as, L1.sa
  for (int c = 0; c < 4; ++c) {
    int cb = convBase[c];
    _Float16* base = pk_conv + c * PK_CONV;
    pack((const float*)d_in[cb + CV_QW],  base + 0 * 65536, HIDDEN, 8);
    pack((const float*)d_in[cb + CV_KW],  base + 1 * 65536, HIDDEN, 8);
    pack((const float*)d_in[cb + CV_VW],  base + 2 * 65536, HIDDEN, 8);
    pack((const float*)d_in[cb + CV_SKW], base + 3 * 65536, HIDDEN, 8);
    pack((const float*)d_in[cb + CV_EW],  base + 4 * 65536, 60, 2);
  }
  pack((const float*)d_in[55], pk_outA, HIDDEN, 8);
  pack((const float*)d_in[57], pk_outS, HIDDEN, 8);

  // ---- input projections + SiLU ------------------------------------------
  launch_gemm(stream, x_agent, pk_inA, (const float*)d_in[10], xa, nullptr, NA, 16, 1, 1);
  launch_gemm(stream, x_scene, pk_inS, (const float*)d_in[12], xs, nullptr, NS, 16, 1, 1);

  // ---- conv runner --------------------------------------------------------
  auto run_conv = [&](int cidx, const float* xsrc, int Nsrc, const float* xdst, int Nd,
                      const int* src, const int* dst, const float* ea, float* outBuf) {
    int cb = convBase[cidx];
    _Float16* base = pk_conv + cidx * PK_CONV;
    launch_gemm(stream, xdst, base + 0 * 65536, (const float*)d_in[cb + CV_QB],
                qbuf, nullptr, Nd, HIDDEN, 8, 0);
    launch_gemm(stream, xsrc, base + 1 * 65536, (const float*)d_in[cb + CV_KB],
                kbuf, nullptr, Nsrc, HIDDEN, 8, 0);
    launch_gemm(stream, xsrc, base + 2 * 65536, (const float*)d_in[cb + CV_VB],
                vbuf, nullptr, Nsrc, HIDDEN, 8, 0);
    // edge projection (no bias) -> f16, K=60 padded to 64
    launch_gemm(stream, ea, base + 4 * 65536, nullptr, nullptr, e16, E, 60, 2, 0);
    // skip projection seeds the output accumulator
    launch_gemm(stream, xdst, base + 3 * 65536, (const float*)d_in[cb + CV_SKB],
                outBuf, nullptr, Nd, HIDDEN, 8, 0);

    (void)hipMemsetAsync(segmax, 0, (size_t)Nd * HEADS * 4, stream);
    (void)hipMemsetAsync(segden, 0, (size_t)Nd * HEADS * 4, stream);

    int eblocks = (E + 7) / 8;   // wave per edge, 8 waves / block
    edge_score_kernel<<<dim3(eblocks), dim3(256), 0, stream>>>(
        qbuf, kbuf, e16, src, dst, score, segmax, E);
    edge_exp_kernel<<<dim3((E * HEADS + 255) / 256), dim3(256), 0, stream>>>(
        score, dst, segmax, segden, E);
    edge_msg_kernel<<<dim3(eblocks), dim3(256), 0, stream>>>(
        score, segden, vbuf, e16, src, dst, outBuf, E);
  };

  const float* norm_a_b = (const float*)d_in[50];
  const float* norm_a_g = (const float*)d_in[51];
  const float* norm_s_b = (const float*)d_in[52];
  const float* norm_s_g = (const float*)d_in[53];

  for (int layer = 0; layer < 2; ++layer) {
    run_conv(layer * 2 + 0, xa, NA, xs, NS, src_as, dst_as, ea_as, new_s);  // agent->scene
    run_conv(layer * 2 + 1, xs, NS, xa, NA, src_sa, dst_sa, ea_sa, new_a);  // scene->agent
    ln_silu_kernel<<<dim3((NA + 7) / 8), dim3(256), 0, stream>>>(new_a, norm_a_g, norm_a_b, xa, NA);
    ln_silu_kernel<<<dim3((NS + 7) / 8), dim3(256), 0, stream>>>(new_s, norm_s_g, norm_s_b, xs, NS);
  }

  // ---- output projections + pooling --------------------------------------
  float* out    = (float*)d_out;
  float* node_a = out;
  float* node_s = out + (size_t)NA * HIDDEN;
  float* pa     = node_s + (size_t)NS * HIDDEN;
  float* ps     = pa + BGRAPH * HIDDEN;
  float* concat = ps + BGRAPH * HIDDEN;

  launch_gemm(stream, xa, pk_outA, (const float*)d_in[54], node_a, nullptr, NA, HIDDEN, 8, 0);
  launch_gemm(stream, xs, pk_outS, (const float*)d_in[56], node_s, nullptr, NS, HIDDEN, 8, 0);

  (void)hipMemsetAsync(pa,  0, (size_t)2 * BGRAPH * HIDDEN * 4, stream);   // pa + ps
  (void)hipMemsetAsync(cnt, 0, 2 * BGRAPH * 4, stream);
  pool_acc_kernel<<<dim3((NA + 7) / 8), dim3(256), 0, stream>>>(node_a, batch_a, pa, cnt, NA);
  pool_acc_kernel<<<dim3((NS + 7) / 8), dim3(256), 0, stream>>>(node_s, batch_s, ps, cnt + BGRAPH, NS);
  pool_final_kernel<<<dim3(BGRAPH * HIDDEN / 256), dim3(256), 0, stream>>>(
      pa, ps, cnt, cnt + BGRAPH, concat);
}
